// LSTM_12979391168897
// MI455X (gfx1250) — compile-verified, matrix operand order checked by reading, not run on hardware
//
#include <hip/hip_runtime.h>
#include <hip/hip_bf16.h>
#include <math.h>

typedef __attribute__((ext_vector_type(16))) _Float16 v16h;
typedef __attribute__((ext_vector_type(8)))  float    v8f;

#define HDIM  128
#define LSEQ  512
#define BATCH 128
#define GATES 512   // 4*H

__device__ __forceinline__ float sigm_(float x) { return 1.0f / (1.0f + __expf(-x)); }

// ---------------------------------------------------------------------------
// Pack [Wih;Whh]^T (K x 4H) into WMMA-B fragment order.
// B fragment (32x16 f16) per (n-tile, k-chunk): element (k_local, n_local):
//   lane = n_local + 16*bg   where bg = ((k_local & 15) >= 8)
//   half = (k_local & 7) + 8*(k_local >> 4)
// Flat: out[((nt*nk + kc)*32 + lane)*16 + half]
// ---------------------------------------------------------------------------
__global__ void pack_weights_kernel(const float* __restrict__ Wih,  // [4H, D]
                                    const float* __restrict__ Whh,  // [4H, H]
                                    int D, int Dp, int nk,
                                    _Float16* __restrict__ out)
{
    const int total = nk * 32 * 512;
    for (int idx = blockIdx.x * blockDim.x + threadIdx.x; idx < total;
         idx += gridDim.x * blockDim.x) {
        int frag = idx >> 9;
        int rem  = idx & 511;
        int lane = rem >> 4;
        int hh   = rem & 15;
        int nt = frag / nk;
        int kc = frag % nk;
        int bg = lane >> 4;
        int nl = lane & 15;
        int kl = (hh & 7) + 8 * bg + 16 * (hh >> 3);
        int k  = kc * 32 + kl;
        int n  = nt * 16 + nl;
        float v;
        if (k < Dp) v = (k < D) ? Wih[n * D + k] : 0.0f;
        else        v = Whh[n * HDIM + (k - Dp)];
        out[idx] = (_Float16)v;
    }
}

// x [B,L,13] f32 -> xh [B,L,32] f16, zero-padded
__global__ void convert_x_kernel(const float* __restrict__ x, _Float16* __restrict__ xh)
{
    int idx = blockIdx.x * blockDim.x + threadIdx.x;  // B*L*32 threads
    int feat = idx & 31;
    int bl   = idx >> 5;
    float v = (feat < 13) ? x[bl * 13 + feat] : 0.0f;
    xh[idx] = (_Float16)v;
}

// ---------------------------------------------------------------------------
// LSTM layer-direction parameters (two sets per launch, blockIdx.z selects,
// so independent layers run concurrently and the recurrent critical path
// shrinks from 5 sequential passes to 3).
// ---------------------------------------------------------------------------
struct LstmP {
    const _Float16* in;      // input activations (f16, feature-stride inStride)
    int             inStride;
    int             Dp;      // padded feature count (multiple of 32)
    const _Float16* wpack;   // packed B fragments, + dir*nk*32*512
    const float*    bih;     // [2, 512]
    const float*    bhh;     // [2, 512]
    _Float16*       out;     // sequence output (ignored in maxMode)
    int             outStride;
    int             outOff0;
    float*          maxOut;  // running-max output (maxMode)
    int             maxOff0;
    int             maxMode;
};

// grid = (B/16, 2 dirs, nLayers); block = 256 (8 wave32).
// Each wave owns 4 N-tiles of the 512 gate columns.
__global__ __launch_bounds__(256, 1) void lstm_dir_kernel(LstmP pa, LstmP pb)
{
    __shared__ alignas(32) _Float16 xA[2][16 * 512]; // double-buffered input A-frags
    __shared__ alignas(32) _Float16 hA[4 * 512];     // h A-frags (K = 128)
    __shared__ float gatesS[16 * GATES];             // [m][n] gate pre-activations

    const LstmP p = (blockIdx.z == 0) ? pa : pb;

    const int tid  = threadIdx.x;
    const int lane = tid & 31;
    const int wave = tid >> 5;
    const int dir  = blockIdx.y;
    const int b0   = blockIdx.x * 16;

    const int nkx = p.Dp >> 5;        // input K-chunks
    const int nk  = nkx + 4;          // + recurrent K-chunks
    const _Float16* wp  = p.wpack + (size_t)dir * nk * 32 * 512;
    const float*    bip = p.bih + dir * 512;
    const float*    bhp = p.bhh + dir * 512;

    // per-wave weight fragment base pointers (time-invariant)
    const int nt0 = wave * 4;
    const _Float16* wq0 = wp + ((size_t)(nt0 + 0) * nk) * 512 + lane * 16;
    const _Float16* wq1 = wp + ((size_t)(nt0 + 1) * nk) * 512 + lane * 16;
    const _Float16* wq2 = wp + ((size_t)(nt0 + 2) * nk) * 512 + lane * 16;
    const _Float16* wq3 = wp + ((size_t)(nt0 + 3) * nk) * 512 + lane * 16;

    // elementwise ownership: fixed j, 8 rows m (time-invariant addresses)
    const int j     = tid & 127;
    const int mbase = (tid >> 7) * 8;
    float creg[8], hmax[8];
    int   gidx[8], hAidx[8];
    size_t outBase[8];
#pragma unroll
    for (int r = 0; r < 8; ++r) {
        const int m = mbase + r;
        creg[r] = 0.0f; hmax[r] = -3.4e38f;
        gidx[r] = m * GATES + j;
        const int chunk = j >> 5;
        const int kl    = j & 31;
        const int ln    = m + 16 * (((kl & 15) >= 8) ? 1 : 0);
        const int hh    = (kl & 7) + 8 * ((kl >> 4) & 1);
        hAidx[r] = chunk * 512 + ln * 16 + hh;
        outBase[r] = (size_t)(b0 + m) * LSEQ * p.outStride + p.outOff0 + dir * 128 + j;
    }
    const float bI = bip[j]       + bhp[j];
    const float bF = bip[128 + j] + bhp[128 + j];
    const float bG = bip[256 + j] + bhp[256 + j];
    const float bO = bip[384 + j] + bhp[384 + j];

    for (int idx = tid; idx < 4 * 512; idx += 256) hA[idx] = (_Float16)0.0f;

    // stage x for t = 0 into buffer 0
    {
        const int ta0 = dir ? (LSEQ - 1) : 0;
        for (int idx = tid; idx < nkx * 512; idx += 256) {
            int chunk = idx >> 9;
            int rem   = idx & 511;
            int ln    = rem >> 4;
            int hh    = rem & 15;
            int m     = ln & 15;
            int bg    = ln >> 4;
            int k     = chunk * 32 + (hh & 7) + 8 * bg + 16 * (hh >> 3);
            xA[0][idx] = p.in[((size_t)(b0 + m) * LSEQ + ta0) * p.inStride + k];
        }
    }
    __syncthreads();

    for (int t = 0; t < LSEQ; ++t) {
        const int ta  = dir ? (LSEQ - 1 - t) : t;
        const int buf = t & 1;

        // gates[16, 512] = [x_t | h] @ [Wih; Whh]^T via WMMA
        v8f acc[4] = { v8f{}, v8f{}, v8f{}, v8f{} };
        {
            const _Float16* xp = &xA[buf][lane * 16];
            for (int kc = 0; kc < nkx; ++kc) {
                v16h a  = *(const v16h*)(xp  + kc * 512);
                v16h b0v = *(const v16h*)(wq0 + kc * 512);
                v16h b1v = *(const v16h*)(wq1 + kc * 512);
                v16h b2v = *(const v16h*)(wq2 + kc * 512);
                v16h b3v = *(const v16h*)(wq3 + kc * 512);
                acc[0] = __builtin_amdgcn_wmma_f32_16x16x32_f16(false, a, false, b0v, (short)0, acc[0], false, false);
                acc[1] = __builtin_amdgcn_wmma_f32_16x16x32_f16(false, a, false, b1v, (short)0, acc[1], false, false);
                acc[2] = __builtin_amdgcn_wmma_f32_16x16x32_f16(false, a, false, b2v, (short)0, acc[2], false, false);
                acc[3] = __builtin_amdgcn_wmma_f32_16x16x32_f16(false, a, false, b3v, (short)0, acc[3], false, false);
            }
        }
#pragma unroll
        for (int kc = 0; kc < 4; ++kc) {
            v16h a = *(const v16h*)&hA[kc * 512 + lane * 16];
            const int kk = nkx + kc;
            v16h b0v = *(const v16h*)(wq0 + (size_t)kk * 512);
            v16h b1v = *(const v16h*)(wq1 + (size_t)kk * 512);
            v16h b2v = *(const v16h*)(wq2 + (size_t)kk * 512);
            v16h b3v = *(const v16h*)(wq3 + (size_t)kk * 512);
            acc[0] = __builtin_amdgcn_wmma_f32_16x16x32_f16(false, a, false, b0v, (short)0, acc[0], false, false);
            acc[1] = __builtin_amdgcn_wmma_f32_16x16x32_f16(false, a, false, b1v, (short)0, acc[1], false, false);
            acc[2] = __builtin_amdgcn_wmma_f32_16x16x32_f16(false, a, false, b2v, (short)0, acc[2], false, false);
            acc[3] = __builtin_amdgcn_wmma_f32_16x16x32_f16(false, a, false, b3v, (short)0, acc[3], false, false);
        }

        // spill C fragments to LDS:  lane 0-15: m=v,   n=lane
        //                            lane16-31: m=v+8, n=lane-16
        {
            const int m0 = (lane >> 4) * 8;
            const int nl = lane & 15;
#pragma unroll
            for (int q = 0; q < 4; ++q)
#pragma unroll
                for (int v = 0; v < 8; ++v)
                    gatesS[(m0 + v) * GATES + (nt0 + q) * 16 + nl] = acc[q][v];
        }
        __syncthreads();

        // elementwise LSTM cell; h staged back to LDS in A-fragment layout
#pragma unroll
        for (int r = 0; r < 8; ++r) {
            const float gi = gatesS[gidx[r]]       + bI;
            const float gf = gatesS[gidx[r] + 128] + bF;
            const float gg = gatesS[gidx[r] + 256] + bG;
            const float go = gatesS[gidx[r] + 384] + bO;
            float c = sigm_(gf) * creg[r] + sigm_(gi) * tanhf(gg);
            creg[r] = c;
            float h = sigm_(go) * tanhf(c);
            hA[hAidx[r]] = (_Float16)h;
            if (p.maxMode) hmax[r] = fmaxf(hmax[r], h);
            else p.out[outBase[r] + (size_t)ta * p.outStride] = (_Float16)h;
        }

        // stage x for t+1 into the other buffer (overlaps elementwise latency)
        if (t + 1 < LSEQ) {
            const int tn = dir ? (LSEQ - 2 - t) : (t + 1);
            _Float16* xw = xA[buf ^ 1];
            for (int idx = tid; idx < nkx * 512; idx += 256) {
                int chunk = idx >> 9;
                int rem   = idx & 511;
                int ln    = rem >> 4;
                int hh    = rem & 15;
                int m     = ln & 15;
                int bg    = ln >> 4;
                int k     = chunk * 32 + (hh & 7) + 8 * bg + 16 * (hh >> 3);
                xw[idx] = p.in[((size_t)(b0 + m) * LSEQ + tn) * p.inStride + k];
            }
        }
        __syncthreads();
    }

    if (p.maxMode) {
#pragma unroll
        for (int r = 0; r < 8; ++r) {
            const int m = mbase + r;
            p.maxOut[(size_t)(b0 + m) * 512 + p.maxOff0 + dir * 128 + j] = hmax[r];
        }
    }
}

// y4 = selu(y3 @ l4_w^T + l4_b)   [128,512]x[512,128]
__global__ void head_l4_kernel(const float* __restrict__ y3, const float* __restrict__ w,
                               const float* __restrict__ b, float* __restrict__ y4)
{
    const int o  = threadIdx.x;   // 128
    const int bb = blockIdx.x;    // 128
    float acc = b[o];
    const float* row = y3 + bb * 512;
    const float* wr  = w + o * 512;
    for (int k = 0; k < 512; ++k) acc += row[k] * wr[k];
    const float scale = 1.0507009873554805f, alpha = 1.6732632423543772f;
    y4[bb * 128 + o] = acc > 0.0f ? scale * acc : scale * alpha * (__expf(acc) - 1.0f);
}

// out = sigmoid(y4 @ fc_w^T + fc_b)   [128,2]
__global__ void head_fc_kernel(const float* __restrict__ y4, const float* __restrict__ w,
                               const float* __restrict__ b, float* __restrict__ outp)
{
    const int t = blockIdx.x * blockDim.x + threadIdx.x;
    if (t >= 256) return;
    const int bb = t >> 1, o = t & 1;
    float acc = b[o];
    for (int k = 0; k < 128; ++k) acc += y4[bb * 128 + k] * w[o * 128 + k];
    outp[t] = sigm_(acc);
}

// ---------------------------------------------------------------------------
extern "C" void kernel_launch(void* const* d_in, const int* in_sizes, int n_in,
                              void* d_out, int out_size, void* d_ws, size_t ws_size,
                              hipStream_t stream)
{
    const float* x     = (const float*)d_in[0];
    const float* Wih1  = (const float*)d_in[1];
    const float* Whh1  = (const float*)d_in[2];
    const float* bih1  = (const float*)d_in[3];
    const float* bhh1  = (const float*)d_in[4];
    const float* Wih21 = (const float*)d_in[5];
    const float* Whh21 = (const float*)d_in[6];
    const float* bih21 = (const float*)d_in[7];
    const float* bhh21 = (const float*)d_in[8];
    const float* Wih22 = (const float*)d_in[9];
    const float* Whh22 = (const float*)d_in[10];
    const float* bih22 = (const float*)d_in[11];
    const float* bhh22 = (const float*)d_in[12];
    const float* Wih31 = (const float*)d_in[13];
    const float* Whh31 = (const float*)d_in[14];
    const float* bih31 = (const float*)d_in[15];
    const float* bhh31 = (const float*)d_in[16];
    const float* Wih32 = (const float*)d_in[17];
    const float* Whh32 = (const float*)d_in[18];
    const float* bih32 = (const float*)d_in[19];
    const float* bhh32 = (const float*)d_in[20];
    const float* l4_w  = (const float*)d_in[21];
    const float* l4_b  = (const float*)d_in[22];
    const float* fc_w  = (const float*)d_in[23];
    const float* fc_b  = (const float*)d_in[24];
    float* outp = (float*)d_out;

    // workspace carve-up
    char* base = (char*)d_ws;
    size_t off = 0;
    auto carve = [&](size_t bytes) { char* p = base + off; off += (bytes + 255) & ~(size_t)255; return p; };

    _Float16* xh = (_Float16*)carve((size_t)BATCH * LSEQ * 32 * 2);    // padded input
    _Float16* y1 = (_Float16*)carve((size_t)BATCH * LSEQ * 256 * 2);   // biLSTM1 out
    _Float16* Y2 = (_Float16*)carve((size_t)BATCH * LSEQ * 512 * 2);   // [y21 | y22]
    float*    mx = (float*)   carve((size_t)BATCH * 512 * 4);          // y3
    float*    y4 = (float*)   carve((size_t)BATCH * 128 * 4);

    const int nk1 = 5, nk21 = 12, nk22 = 5, nk31 = 20, nk32 = 12;
    _Float16* wp1  = (_Float16*)carve((size_t)2 * nk1  * 32 * 512 * 2);
    _Float16* wp21 = (_Float16*)carve((size_t)2 * nk21 * 32 * 512 * 2);
    _Float16* wp22 = (_Float16*)carve((size_t)2 * nk22 * 32 * 512 * 2);
    _Float16* wp31 = (_Float16*)carve((size_t)2 * nk31 * 32 * 512 * 2);
    _Float16* wp32 = (_Float16*)carve((size_t)2 * nk32 * 32 * 512 * 2);
    (void)ws_size; (void)n_in; (void)in_sizes; (void)out_size;

    // pack weights for each layer x direction
    auto pack = [&](const float* Wih, const float* Whh, int D, int Dp, int nk, _Float16* wp) {
        const int total = nk * 32 * 512;
        const int grid  = (total + 255) / 256;
        for (int d = 0; d < 2; ++d) {
            pack_weights_kernel<<<grid, 256, 0, stream>>>(
                Wih + (size_t)d * GATES * D, Whh + (size_t)d * GATES * HDIM,
                D, Dp, nk, wp + (size_t)d * nk * 32 * 512);
        }
    };
    pack(Wih1,  Whh1,  13,  32,  nk1,  wp1);
    pack(Wih21, Whh21, 256, 256, nk21, wp21);
    pack(Wih22, Whh22, 13,  32,  nk22, wp22);
    pack(Wih31, Whh31, 512, 512, nk31, wp31);
    pack(Wih32, Whh32, 256, 256, nk32, wp32);

    convert_x_kernel<<<(BATCH * LSEQ * 32) / 256, 256, 0, stream>>>(x, xh);

    // layer param sets
    LstmP P1  = { xh, 32, 32,   wp1,  bih1,  bhh1,  y1, 256, 0,   mx, 0,   0 };
    LstmP P22 = { xh, 32, 32,   wp22, bih22, bhh22, Y2, 512, 256, mx, 0,   0 };
    LstmP P21 = { y1, 256, 256, wp21, bih21, bhh21, Y2, 512, 0,   mx, 0,   0 };
    LstmP P31 = { Y2, 512, 512, wp31, bih31, bhh31, Y2, 512, 0,   mx, 0,   1 };
    LstmP P32 = { Y2, 512, 256, wp32, bih32, bhh32, Y2, 512, 0,   mx, 256, 1 };

    // pass A: layer1 || layer22 (both read xh, independent)
    lstm_dir_kernel<<<dim3(BATCH / 16, 2, 2), 256, 0, stream>>>(P1, P22);
    // pass B: layer21 (needs y1)
    lstm_dir_kernel<<<dim3(BATCH / 16, 2, 1), 256, 0, stream>>>(P21, P21);
    // pass C: layer31 || layer32 (both read Y2, fused max-over-time)
    lstm_dir_kernel<<<dim3(BATCH / 16, 2, 2), 256, 0, stream>>>(P31, P32);

    head_l4_kernel<<<BATCH, 128, 0, stream>>>(mx, l4_w, l4_b, y4);
    head_fc_kernel<<<1, 256, 0, stream>>>(y4, fc_w, fc_b, outp);
}
